// CcLoss_18708877541513
// MI455X (gfx1250) — matrix operand order repaired and checked by reading, not compiled, not run on hardware
//
#include <hip/hip_runtime.h>
#include <hip/hip_bf16.h>
#include <math.h>

// Problem constants (from reference): B=64, P=1024, D=128
#define BB 64
#define PP 1024
#define DD 128
#define QT 32               // q rows staged per iteration
#define NT (PP / QT)        // 32 iterations

typedef __attribute__((ext_vector_type(16))) __bf16 v16bf;
typedef __attribute__((ext_vector_type(8)))  __bf16 v8bf;
typedef __attribute__((ext_vector_type(8)))  float  v8f;
typedef __attribute__((ext_vector_type(4))) unsigned int u32x4;
typedef __attribute__((ext_vector_type(8))) int i32x8;
typedef __attribute__((ext_vector_type(4))) int i32x4;

// Device pass (gfx1250) takes the TDM path; host pass (no __gfx1250__) parses the fallback.
#if defined(__gfx1250__) && __has_builtin(__builtin_amdgcn_tensor_load_to_lds)
#define USE_TDM 1
#else
#define USE_TDM 0
#endif

__device__ __forceinline__ float atomic_add_f32(float* p, float v) {
  return __hip_atomic_fetch_add(p, v, __ATOMIC_RELAXED, __HIP_MEMORY_SCOPE_AGENT);
}

#if USE_TDM
// Issue one TDM 2-D tile load: 32 rows x 128 f32 (row stride 128 f32) -> LDS.
// D# per CDNA5 ISA §8.3/8.4. Group2/3 zero (<=2D tensor).
__device__ __forceinline__ void tdm_load_tile(const float* gsrc, void* lds_dst) {
  unsigned lds_off = (unsigned)(size_t)lds_dst;             // flat LDS addr low 32 = offset
  unsigned long long ga = (unsigned long long)(size_t)gsrc;
  u32x4 g0;
  g0[0] = 1u;                                               // count=1, user descriptor
  g0[1] = lds_off;                                          // lds_addr
  g0[2] = (unsigned)ga;                                     // global_addr[31:0]
  g0[3] = (unsigned)((ga >> 32) & 0x01FFFFFFu) | (2u << 30);// global_addr[56:32] | type=2
  i32x8 g1;
  g1[0] = 0x20000;                  // workgroup_mask=0 | data_size=2 (4 bytes)
  g1[1] = (int)(128u << 16);        // tensor_dim0 = 128 elems (bits 79:48 low half)
  g1[2] = (int)((unsigned)QT << 16);// tensor_dim0 hi=0 | tensor_dim1 = 32
  g1[3] = (int)(128u << 16);        // tensor_dim1 hi=0 | tile_dim0 = 128
  g1[4] = QT;                       // tile_dim1 = 32 | tile_dim2 = 0
  g1[5] = DD;                       // tensor_dim0_stride = 128 elems
  g1[6] = 0;
  g1[7] = 0;                        // tensor_dim1_stride = 0 (2-D tile)
  i32x4 z4 = {0, 0, 0, 0};
#if __clang_major__ >= 23
  i32x8 z8 = {0, 0, 0, 0, 0, 0, 0, 0};
  __builtin_amdgcn_tensor_load_to_lds(g0, g1, z4, z4, z8, 0);
#else
  __builtin_amdgcn_tensor_load_to_lds(g0, g1, z4, z4, 0);
#endif
}
#endif

// ---------------- kernel 0: zero accumulators ----------------
__global__ void zero_kernel(float* p, int n) {
  int i = blockIdx.x * blockDim.x + threadIdx.x;
  if (i < n) p[i] = 0.0f;
}

// ---------------- kernel 1: per-row inverse L2 norms ----------------
__global__ __launch_bounds__(256) void rnorm_kernel(const float* __restrict__ feat,
                                                    float* __restrict__ rnorm) {
  int row = blockIdx.x * 256 + threadIdx.x;   // 0 .. B*P-1
  const float4* fr = (const float4*)(feat + (size_t)row * DD);
  float s = 0.0f;
#pragma unroll
  for (int i = 0; i < DD / 4; ++i) {
    float4 v = fr[i];
    s += v.x * v.x + v.y * v.y + v.z * v.z + v.w * v.w;
  }
  rnorm[row] = rsqrtf(s);
}

// ---------------- kernel 2: fused sim -> mask -> proto -> stats ----------------
// grid: B * (P/64) blocks, 128 threads (4 waves); each wave owns 16 p-rows.
__global__ __launch_bounds__(128) void cc_main_kernel(
    const float* __restrict__ feat, const float* __restrict__ tau_p,
    const float* __restrict__ rnorm,
    float* __restrict__ T,   // [B*128] per-d column sums of proto
    float* __restrict__ S1,  // [B]
    float* __restrict__ S2,  // [B]
    float* __restrict__ MSE) // [1]
{
#if USE_TDM
  __shared__ alignas(16) float f32tile[2][QT * DD];     // 32 KB TDM double buffer
#endif
  __shared__ alignas(32) __bf16 qmaj[QT * DD];          // 8 KB: [q][d] for sim B-frags
  __shared__ alignas(32) __bf16 dmaj[DD * QT];          // 8 KB: [d][q] for proto B-frags
  __shared__ alignas(16) __bf16 lds_mask[4][16 * QT];   // 4 KB per-wave mask tiles

  const int tid  = threadIdx.x;
  const int lane = tid & 31;
  const int wv   = tid >> 5;
  const int b    = blockIdx.x >> 4;          // 16 blocks per batch
  const int p0   = (blockIdx.x & 15) * 64;   // this block's 64 p-rows
  const int m    = lane & 15;
  const int h    = lane >> 4;                // lane half (0/1)
  const float tau = tau_p[0];

  const float* fb = feat + (size_t)b * PP * DD;
  const float* rb = rnorm + b * PP;

  // --- A fragments (16 p-rows x 128 dims) from global, f32 -> bf16.
  // A layout: halves[0..15] = K in [16h, 16h+16)
  const float* Arow = fb + (size_t)(p0 + wv * 16 + m) * DD;
  v16bf afrag[4];
#pragma unroll
  for (int f = 0; f < 4; ++f) {
    const float* s0 = Arow + f * 32 + 16 * h;
    const float* s1 = Arow + f * 32 + 16 * h + 8;
#pragma unroll
    for (int i = 0; i < 8; ++i) {
      afrag[f][i]     = (__bf16)s0[i];
      afrag[f][8 + i] = (__bf16)s1[i];
    }
  }

  float rp[8];
#pragma unroll
  for (int r = 0; r < 8; ++r) rp[r] = rb[p0 + wv * 16 + r + 8 * h];

  v8f acc[8];
#pragma unroll
  for (int n = 0; n < 8; ++n) acc[n] = {};
  float deg[8];
#pragma unroll
  for (int r = 0; r < 8; ++r) deg[r] = 0.0f;

  // convert-stage ownership: one (q-row, 32-d chunk) per thread
  const int qq    = tid >> 2;
  const int dbase = (tid & 3) * 32;

#if USE_TDM
  if (wv == 0) tdm_load_tile(fb, &f32tile[0][0]);   // prefetch tile 0
#endif

  for (int t = 0; t < NT; ++t) {
    const int q0 = t * QT;
#if USE_TDM
    if (wv == 0) __builtin_amdgcn_s_wait_tensorcnt((short)0);  // tile t landed
#endif
    __syncthreads();  // tile t visible; prior compute done reading qmaj/dmaj

    // --- convert f32 tile -> bf16 in two layouts
#if USE_TDM
    const float* src = &f32tile[t & 1][qq * DD + dbase];
#else
    const float* src = fb + (size_t)(q0 + qq) * DD + dbase;
#endif
    {
      __bf16 tmp[32];
      const float4* s4 = (const float4*)src;
#pragma unroll
      for (int c = 0; c < 8; ++c) {
        float4 v = s4[c];
        tmp[c * 4 + 0] = (__bf16)v.x;
        tmp[c * 4 + 1] = (__bf16)v.y;
        tmp[c * 4 + 2] = (__bf16)v.z;
        tmp[c * 4 + 3] = (__bf16)v.w;
      }
#pragma unroll
      for (int c = 0; c < 4; ++c) {          // q-major: 4x 16B vector stores
        v8bf pk;
#pragma unroll
        for (int i = 0; i < 8; ++i) pk[i] = tmp[c * 8 + i];
        *(v8bf*)&qmaj[qq * DD + dbase + c * 8] = pk;
      }
#pragma unroll
      for (int i = 0; i < 32; ++i)           // d-major scatter
        dmaj[(dbase + i) * QT + qq] = tmp[i];
    }
    __syncthreads();  // conversions visible

#if USE_TDM
    if (wv == 0 && t + 1 < NT)               // overlap DMA of tile t+1 with compute
      tdm_load_tile(fb + (size_t)(q0 + QT) * DD, &f32tile[(t + 1) & 1][0]);
#endif

    // --- sim: two 16x16 q sub-tiles, K=128 via 4 chained bf16 WMMAs each
#pragma unroll
    for (int s = 0; s < 2; ++s) {
      v8f c = {};
#pragma unroll
      for (int f = 0; f < 4; ++f) {
        v16bf bf = *(const v16bf*)&qmaj[(s * 16 + m) * DD + f * 32 + 16 * h];
        c = __builtin_amdgcn_wmma_f32_16x16x32_bf16(false, afrag[f], false, bf,
                                                    (short)0, c, false, false);
      }
      float rq = rb[q0 + s * 16 + m];
#pragma unroll
      for (int r = 0; r < 8; ++r) {
        float sv = c[r] * rp[r] * rq;            // cosine similarity
        float mk = (sv > tau) ? 1.0f : 0.0f;     // adjacency mask
        deg[r] += mk;
        lds_mask[wv][(r + 8 * h) * QT + s * 16 + m] = (__bf16)mk;
      }
    }
    __syncthreads();  // mask tiles visible (also orders vs next-iter overwrites)

    // --- proto += mask(16x32) @ featq(32x128): contiguous frags from lds
    v8bf lo = *(const v8bf*)&lds_mask[wv][m * QT + 16 * h];
    v8bf hi = *(const v8bf*)&lds_mask[wv][m * QT + 16 * h + 8];
    v16bf mfrag;
#pragma unroll
    for (int i = 0; i < 8; ++i) { mfrag[i] = lo[i]; mfrag[8 + i] = hi[i]; }
#pragma unroll
    for (int n = 0; n < 8; ++n) {
      v16bf bf = *(const v16bf*)&dmaj[(n * 16 + m) * QT + 16 * h];
      acc[n] = __builtin_amdgcn_wmma_f32_16x16x32_bf16(false, mfrag, false, bf,
                                                       (short)0, acc[n], false, false);
    }
  }

  // --- deg: reduce across the 16 lanes of each half
#pragma unroll
  for (int r = 0; r < 8; ++r) {
    float d = deg[r];
#pragma unroll
    for (int off = 1; off < 16; off <<= 1) d += __shfl_xor(d, off, 32);
    deg[r] = d;   // every lane in a half has deg for M = r + 8*h
  }

  // --- normalize, accumulate statistics
  float s1 = 0.0f, s2 = 0.0f, mse = 0.0f;
  float t8[8];
#pragma unroll
  for (int n = 0; n < 8; ++n) {
    float tn = 0.0f;
#pragma unroll
    for (int r = 0; r < 8; ++r) {
      float pv = acc[n][r] / deg[r];  // deg >= 1 (diagonal passes threshold)
      float fv = fb[(size_t)(p0 + wv * 16 + r + 8 * h) * DD + n * 16 + m];
      float df = pv - fv;
      s1 += pv; s2 += pv * pv; mse += df * df; tn += pv;
    }
    t8[n] = tn + __shfl_xor(tn, 16, 32);  // fold lane halves (same d)
  }
  if (lane < 16) {
#pragma unroll
    for (int n = 0; n < 8; ++n)
      atomic_add_f32(&T[b * DD + n * 16 + lane], t8[n]);
  }
#pragma unroll
  for (int off = 1; off < 32; off <<= 1) {
    s1  += __shfl_xor(s1, off, 32);
    s2  += __shfl_xor(s2, off, 32);
    mse += __shfl_xor(mse, off, 32);
  }
  if (lane == 0) {
    atomic_add_f32(&S1[b], s1);
    atomic_add_f32(&S2[b], s2);
    atomic_add_f32(MSE, mse);
  }
}

// ---------------- kernel 3: closed-form Pearson + final scalar ----------------
// corr[b] = sqrt( P * (sum_d T^2 / P^2 - D*mu^2) / (S2 - P*D*mu^2) )
__global__ __launch_bounds__(64) void final_kernel(
    const float* __restrict__ T, const float* __restrict__ S1,
    const float* __restrict__ S2, const float* __restrict__ MSE,
    float* __restrict__ out)
{
  __shared__ float sh[BB];
  int b = threadIdx.x;
  const float N = (float)PP * (float)DD;
  float mu = S1[b] / N;
  float varX = fmaxf(S2[b] - N * mu * mu, 1e-30f);
  float sT2 = 0.0f;
  for (int d = 0; d < DD; ++d) { float td = T[b * DD + d]; sT2 += td * td; }
  float A = sT2 / ((float)PP * (float)PP) - (float)DD * mu * mu;
  float corr = sqrtf(fmaxf((float)PP * A, 0.0f) / varX);
  sh[b] = 0.5f * (corr + 1.0f);
  __syncthreads();
  if (b == 0) {
    float s = 0.0f;
    for (int i = 0; i < BB; ++i) s += sh[i];
    out[0] = MSE[0] / ((float)BB * N) + s / (float)BB;
  }
}

// ---------------- host launcher ----------------
extern "C" void kernel_launch(void* const* d_in, const int* in_sizes, int n_in,
                              void* d_out, int out_size, void* d_ws, size_t ws_size,
                              hipStream_t stream) {
  const float* feat = (const float*)d_in[0];   // [B,P,D] f32
  const float* tau  = (const float*)d_in[1];   // scalar f32

  float* ws    = (float*)d_ws;
  float* rnorm = ws;                 // B*P  = 65536 floats
  float* T     = ws + BB * PP;       // B*D  =  8192 floats
  float* S1    = T + BB * DD;        // B
  float* S2    = S1 + BB;            // B
  float* MSE   = S2 + BB;            // 1
  const int n_zero = BB * DD + BB + BB + 1;    // zero T,S1,S2,MSE each call

  zero_kernel<<<(n_zero + 255) / 256, 256, 0, stream>>>(T, n_zero);
  rnorm_kernel<<<(BB * PP) / 256, 256, 0, stream>>>(feat, rnorm);
  cc_main_kernel<<<BB * (PP / 64), 128, 0, stream>>>(feat, tau, rnorm, T, S1, S2, MSE);
  final_kernel<<<1, BB, 0, stream>>>(T, S1, S2, MSE, (float*)d_out);
}
